// CADGroupingGNN_10067403342134
// MI455X (gfx1250) — matrix operand order repaired and checked by reading, not compiled, not run on hardware
//
#include <hip/hip_runtime.h>
#include <hip/hip_bf16.h>
#include <math.h>

// ---------------------------------------------------------------------------
// GCN on gfx1250: bf16 WMMA GEMMs (multi-tile waves, pre-packed B fragments)
// + atomic scatter aggregation.
// ---------------------------------------------------------------------------

typedef __attribute__((ext_vector_type(16))) __bf16 v16bf;
typedef __attribute__((ext_vector_type(8)))  float  v8f;

// ---------------- degree / dinv ----------------
__global__ void deg_init_kernel(float* __restrict__ deg, int n) {
  int i = blockIdx.x * blockDim.x + threadIdx.x;
  if (i < n) deg[i] = 1.0f;  // self-loop
}

__global__ void deg_count_kernel(const long long* __restrict__ ei,
                                 float* __restrict__ deg, int E) {
  int e = blockIdx.x * blockDim.x + threadIdx.x;
  if (e < E) atomicAdd(&deg[(int)ei[(size_t)E + e]], 1.0f);
}

__global__ void deg_rsqrt_kernel(float* __restrict__ deg, int n) {
  int i = blockIdx.x * blockDim.x + threadIdx.x;
  if (i < n) deg[i] = rsqrtf(deg[i]);  // deg >= 1, safe
}

// ---------------- pack W[K, ncols] f32 -> WMMA B-fragment bf16 layout -------
// Packed index = ((colTile*(K/32) + kChunk)*32 + lane)*16 + i, where the lane
// holds column colTile*16 + (lane&15) and K-row kChunk*32 + (lane>>4)*8 + (i&7) + (i>>3)*16.
// GEMM then reads one contiguous v16bf (32 B) per lane per K-chunk per tile.
__global__ void pack_w_kernel(const float* __restrict__ W,
                              __bf16* __restrict__ Wp, int K, int ncols) {
  int idx = blockIdx.x * blockDim.x + threadIdx.x;
  if (idx >= K * ncols) return;
  const int i    = idx & 15;
  const int lane = (idx >> 4) & 31;
  const int rest = idx >> 9;
  const int nkc  = K >> 5;
  const int c    = rest % nkc;  // k-chunk
  const int t    = rest / nkc;  // column tile
  const int kk   = c * 32 + ((lane >> 4) * 8) + (i & 7) + ((i >> 3) * 16);
  const int col  = t * 16 + (lane & 15);
  Wp[idx] = (__bf16)W[(size_t)kk * ncols + col];
}

// ---------------- WMMA GEMM: Y[n,NCOLS] = act(X[n,128] @ W + bias) ----------
// ACT: 0 none, 1 relu, 2 sigmoid.
// Each wave owns one 16-row A fragment and TPW column tiles: A is loaded and
// converted once per K-chunk, reused by TPW WMMAs (A-redundancy 8x -> 2x).
// NCOLS=128: 8 waves = 4 row tiles x 2 col groups (64 rows/block), 16 WMMAs.
// NCOLS=32 : 8 waves = 8 row tiles x 1 col group (128 rows/block),  8 WMMAs.
template <int NCOLS, int ACT, bool BIAS>
__global__ void gemm_wmma_bf16_kernel(const float* __restrict__ X,
                                      const __bf16* __restrict__ Wp,
                                      const float* __restrict__ bias,
                                      float* __restrict__ Y, int n) {
  constexpr int K   = 128;
  constexpr int CT  = NCOLS / 16;          // total column tiles (8 or 2)
  constexpr int TPW = (CT >= 4) ? 4 : CT;  // column tiles per wave (4 or 2)
  constexpr int NH  = CT / TPW;            // column groups (2 or 1)
  constexpr int RPB = (8 / NH) * 16;       // rows per block (64 or 128)

  const int lane    = threadIdx.x & 31;
  const int wave    = threadIdx.x >> 5;
  const int rowTile = wave / NH;
  const int colGrp  = wave % NH;
  const int rowBase = blockIdx.x * RPB + rowTile * 16;

  int aRow = rowBase + (lane & 15);
  if (aRow >= n) aRow = n - 1;           // clamp loads; stores masked below
  const int     kHalf = (lane >> 4) * 8; // A K-offset: 0 or 8
  const float4* x4    = (const float4*)(X + (size_t)aRow * K);
  const v16bf*  Wv    = (const v16bf*)Wp;

  v8f acc[TPW] = {};
#pragma unroll
  for (int c = 0; c < K / 32; ++c) {
    // A fragment: K = c*32 + {kHalf..kHalf+7, kHalf+16..kHalf+23}
    const int q  = (c * 32 + kHalf) >> 2;  // float4 index (kHalf % 4 == 0)
    const float4 p0 = x4[q];
    const float4 p1 = x4[q + 1];
    const float4 p2 = x4[q + 4];
    const float4 p3 = x4[q + 5];
    v16bf a;
    a[0]  = (__bf16)p0.x; a[1]  = (__bf16)p0.y; a[2]  = (__bf16)p0.z; a[3]  = (__bf16)p0.w;
    a[4]  = (__bf16)p1.x; a[5]  = (__bf16)p1.y; a[6]  = (__bf16)p1.z; a[7]  = (__bf16)p1.w;
    a[8]  = (__bf16)p2.x; a[9]  = (__bf16)p2.y; a[10] = (__bf16)p2.z; a[11] = (__bf16)p2.w;
    a[12] = (__bf16)p3.x; a[13] = (__bf16)p3.y; a[14] = (__bf16)p3.z; a[15] = (__bf16)p3.w;
#pragma unroll
    for (int t = 0; t < TPW; ++t) {
      const int tile = colGrp * TPW + t;
      const v16bf b  = Wv[(tile * (K / 32) + c) * 32 + lane];  // 32 B vector
      acc[t] = __builtin_amdgcn_wmma_f32_16x16x32_bf16(
          false, a, false, b, (short)0, acc[t], false, false);
    }
  }

  // Epilogue: D VGPR r -> row (lane>=16 ? 8 : 0)+r, col = lane&15
  const int mBase = (lane >> 4) * 8;
#pragma unroll
  for (int t = 0; t < TPW; ++t) {
    const int   col = (colGrp * TPW + t) * 16 + (lane & 15);
    const float bv  = BIAS ? bias[col] : 0.0f;
#pragma unroll
    for (int r = 0; r < 8; ++r) {
      const int row = rowBase + mBase + r;
      if (row < n) {
        float v = acc[t][r] + bv;
        if (ACT == 1) v = fmaxf(v, 0.0f);
        if (ACT == 2) v = __builtin_amdgcn_rcpf(1.0f + __expf(-v));
        Y[(size_t)row * NCOLS + col] = v;
      }
    }
  }
}

// ---------------- self-loop init: agg = h * dinv^2 (C == 128) ----------------
__global__ void self_init_kernel(const float* __restrict__ h,
                                 const float* __restrict__ dinv,
                                 float* __restrict__ agg, int total) {
  int i = blockIdx.x * blockDim.x + threadIdx.x;
  if (i < total) {
    const float di = dinv[i >> 7];  // node = i / 128
    agg[i] = h[i] * di * di;
  }
}

// ---------------- edge scatter: one wave per edge, float4 per lane ----------
__global__ void edge_scatter_kernel(const long long* __restrict__ ei,
                                    const float* __restrict__ h,
                                    const float* __restrict__ dinv,
                                    float* __restrict__ agg, int E) {
  const int t = blockIdx.x * blockDim.x + threadIdx.x;
  const int e = t >> 5;  // 32 lanes cover 128 channels (float4 each)
  if (e >= E) return;
  const int lane = t & 31;
  const int src = (int)ei[e];
  const int dst = (int)ei[(size_t)E + e];
  const float w = dinv[src] * dinv[dst];
  const float4 v = ((const float4*)(h + (size_t)src * 128))[lane];
  float* ad = agg + (size_t)dst * 128 + lane * 4;
  atomicAdd(ad + 0, v.x * w);
  atomicAdd(ad + 1, v.y * w);
  atomicAdd(ad + 2, v.z * w);
  atomicAdd(ad + 3, v.w * w);
}

// ---------------- bias + relu in place (C == 128) ----------------
__global__ void bias_relu_kernel(float* __restrict__ agg,
                                 const float* __restrict__ b, int total) {
  int i = blockIdx.x * blockDim.x + threadIdx.x;
  if (i < total) agg[i] = fmaxf(agg[i] + b[i & 127], 0.0f);
}

// ---------------------------------------------------------------------------
extern "C" void kernel_launch(void* const* d_in, const int* in_sizes, int n_in,
                              void* d_out, int out_size, void* d_ws, size_t ws_size,
                              hipStream_t stream) {
  const float*     x   = (const float*)d_in[0];
  const long long* ei  = (const long long*)d_in[1];
  // d_in[2] = batch (unused by the reference computation)
  const float* W0  = (const float*)d_in[3];
  const float* b0  = (const float*)d_in[4];
  const float* W1  = (const float*)d_in[5];
  const float* b1  = (const float*)d_in[6];
  const float* W2  = (const float*)d_in[7];
  const float* b2  = (const float*)d_in[8];
  const float* Wc1 = (const float*)d_in[9];
  const float* bc1 = (const float*)d_in[10];
  const float* Wc2 = (const float*)d_in[11];
  const float* bc2 = (const float*)d_in[12];
  float* out = (float*)d_out;

  const int C  = 128;
  const int OC = 32;
  const int N  = in_sizes[0] / C;   // 100000
  const int E  = in_sizes[1] / 2;   // 1600000

  // ---- workspace partition ----
  char*  base = (char*)d_ws;
  size_t off  = 0;
  auto alloc = [&](size_t bytes) -> void* {
    void* p = base + off;
    off += (bytes + 255) & ~(size_t)255;
    return p;
  };
  float*  bufA = (float*)alloc((size_t)N * C * sizeof(float));
  float*  bufB = (float*)alloc((size_t)N * C * sizeof(float));
  float*  dinv = (float*)alloc((size_t)N * sizeof(float));
  __bf16* wb0  = (__bf16*)alloc((size_t)C * C  * sizeof(__bf16));
  __bf16* wb1  = (__bf16*)alloc((size_t)C * C  * sizeof(__bf16));
  __bf16* wb2  = (__bf16*)alloc((size_t)C * C  * sizeof(__bf16));
  __bf16* wbc1 = (__bf16*)alloc((size_t)C * C  * sizeof(__bf16));
  __bf16* wbc2 = (__bf16*)alloc((size_t)C * OC * sizeof(__bf16));
  (void)ws_size; (void)n_in; (void)out_size;

  const int TB = 256;
  auto blocks = [&](long long work) { return (unsigned)((work + TB - 1) / TB); };

  // ---- pack weights into WMMA B-fragment bf16 layout (deterministic) ----
  pack_w_kernel<<<blocks(C * C), TB, 0, stream>>>(W0,  wb0,  C, C);
  pack_w_kernel<<<blocks(C * C), TB, 0, stream>>>(W1,  wb1,  C, C);
  pack_w_kernel<<<blocks(C * C), TB, 0, stream>>>(W2,  wb2,  C, C);
  pack_w_kernel<<<blocks(C * C), TB, 0, stream>>>(Wc1, wbc1, C, C);
  pack_w_kernel<<<blocks(C * OC), TB, 0, stream>>>(Wc2, wbc2, C, OC);

  // ---- degree -> dinv (dst-only: compute once, reuse for all 3 layers) ----
  deg_init_kernel <<<blocks(N), TB, 0, stream>>>(dinv, N);
  deg_count_kernel<<<blocks(E), TB, 0, stream>>>(ei, dinv, E);
  deg_rsqrt_kernel<<<blocks(N), TB, 0, stream>>>(dinv, N);

  const long long NC = (long long)N * C;
  const unsigned g128 = (unsigned)((N + 63) / 64);    // RPB=64 for NCOLS=128
  const unsigned g32  = (unsigned)((N + 127) / 128);  // RPB=128 for NCOLS=32
  const long long edgeWork = (long long)E * 32;

  // ---- GCN layer 0: x -> bufB ----
  gemm_wmma_bf16_kernel<128, 0, false><<<g128, TB, 0, stream>>>(x, wb0, nullptr, bufA, N);
  self_init_kernel   <<<blocks(NC), TB, 0, stream>>>(bufA, dinv, bufB, (int)NC);
  edge_scatter_kernel<<<blocks(edgeWork), TB, 0, stream>>>(ei, bufA, dinv, bufB, E);
  bias_relu_kernel   <<<blocks(NC), TB, 0, stream>>>(bufB, b0, (int)NC);

  // ---- GCN layer 1: bufB -> bufB ----
  gemm_wmma_bf16_kernel<128, 0, false><<<g128, TB, 0, stream>>>(bufB, wb1, nullptr, bufA, N);
  self_init_kernel   <<<blocks(NC), TB, 0, stream>>>(bufA, dinv, bufB, (int)NC);
  edge_scatter_kernel<<<blocks(edgeWork), TB, 0, stream>>>(ei, bufA, dinv, bufB, E);
  bias_relu_kernel   <<<blocks(NC), TB, 0, stream>>>(bufB, b1, (int)NC);

  // ---- GCN layer 2: bufB -> bufB ----
  gemm_wmma_bf16_kernel<128, 0, false><<<g128, TB, 0, stream>>>(bufB, wb2, nullptr, bufA, N);
  self_init_kernel   <<<blocks(NC), TB, 0, stream>>>(bufA, dinv, bufB, (int)NC);
  edge_scatter_kernel<<<blocks(edgeWork), TB, 0, stream>>>(ei, bufA, dinv, bufB, E);
  bias_relu_kernel   <<<blocks(NC), TB, 0, stream>>>(bufB, b2, (int)NC);

  // ---- classifier ----
  gemm_wmma_bf16_kernel<128, 1, true><<<g128, TB, 0, stream>>>(bufB, wbc1, bc1, bufA, N);
  gemm_wmma_bf16_kernel<32,  2, true><<<g32,  TB, 0, stream>>>(bufA, wbc2, bc2, out,  N);
}
